// MultiHeadAttention_60473139527922
// MI455X (gfx1250) — matrix operand order repaired and checked by reading, not compiled, not run on hardware
//
#include <hip/hip_runtime.h>

typedef __attribute__((ext_vector_type(16))) __bf16 v16bf;
typedef __attribute__((ext_vector_type(8)))  float  v8f;
typedef __attribute__((ext_vector_type(8)))  unsigned short ush8;
typedef __attribute__((ext_vector_type(16))) unsigned short ush16;
typedef __attribute__((ext_vector_type(4)))  int    v4i;

#define B_  2
#define T_  2048
#define C_  2048
#define NH_ 16
#define NKV_ 4
#define HD_ 128
#define KV_ 512           // NKV_*HD_
#define M_  (B_ * T_)     // 4096

#define LDA 40            // padded LDS row stride (halves): 20 dwords, conflict-free
#define LDK 136           // padded K-tile row stride (halves): 68 dwords, conflict-free

#if defined(__has_builtin)
#if __has_builtin(__builtin_amdgcn_global_load_async_to_lds_b128)
#define ASYNC_OK 1
#endif
#endif

__device__ __forceinline__ unsigned short f2bf(float f) {
  unsigned u = __builtin_bit_cast(unsigned, f);
  u = (u + 0x7FFFu + ((u >> 16) & 1u)) >> 16;   // round-to-nearest-even
  return (unsigned short)u;
}
__device__ __forceinline__ float bf2f(unsigned short h) {
  return __builtin_bit_cast(float, ((unsigned)h) << 16);
}

// ---- async LDS staging (GLOBAL_LOAD_ASYNC_TO_LDS_B128, ASYNCcnt) ----------
__device__ __forceinline__ void async_ld16(unsigned short* lds, const unsigned short* g) {
#if defined(ASYNC_OK)
  __builtin_amdgcn_global_load_async_to_lds_b128(
      (__attribute__((address_space(1))) v4i*)g,
      (__attribute__((address_space(3))) v4i*)lds, 0, 0);
#else
  *(ush8*)lds = *(const ush8*)g;
#endif
}
__device__ __forceinline__ void wait_async() {
#if defined(ASYNC_OK)
#if __has_builtin(__builtin_amdgcn_s_wait_asynccnt)
  __builtin_amdgcn_s_wait_asynccnt(0);
#else
  asm volatile("s_wait_asynccnt 0x0" ::: "memory");
#endif
#endif
}

// ---- WMMA fragment loaders (wave32, V_WMMA_F32_16X16X32_BF16) -------------
// A 16x32 (16-bit): lanes 0-15 row=lane, K{0..7,16..23}; lanes 16-31 row=lane-16,
// K{8..15,24..31}. Two b128 loads per lane.
__device__ __forceinline__ v16bf load_a16x32(const unsigned short* base, int ld) {
  int lane = threadIdx.x & 31;
  int r = lane & 15, h = lane >> 4;
  ush8 c0 = *(const ush8*)(base + (size_t)r * ld + h * 8);
  ush8 c1 = *(const ush8*)(base + (size_t)r * ld + 16 + h * 8);
  ush16 t;
#pragma unroll
  for (int i = 0; i < 8; ++i) { t[i] = c0[i]; t[i + 8] = c1[i]; }
  return __builtin_bit_cast(v16bf, t);
}
// B 32x16 (16-bit): lanes 0-15 col=lane with K{0..7,16..23}; lanes 16-31 col=lane-16,
// K{8..15,24..31}. Source is "B^T" row-major (N x K).
__device__ __forceinline__ v16bf load_b32x16(const unsigned short* base, int ld) {
  int lane = threadIdx.x & 31;
  int n = lane & 15, h = lane >> 4;
  const unsigned short* p = base + (size_t)n * ld + h * 8;
  ush8 c0 = *(const ush8*)(p);
  ush8 c1 = *(const ush8*)(p + 16);
  ush16 t;
#pragma unroll
  for (int i = 0; i < 8; ++i) { t[i] = c0[i]; t[i + 8] = c1[i]; }
  return __builtin_bit_cast(v16bf, t);
}
__device__ __forceinline__ v8f wmma_bf16(v16bf a, v16bf b, v8f c) {
  return __builtin_amdgcn_wmma_f32_16x16x32_bf16(false, a, false, b, (short)0, c,
                                                 false, false);
}

// ---- conversion kernels ----------------------------------------------------
__global__ __launch_bounds__(256) void cvt_bf16(const float* __restrict__ s,
                                                unsigned short* __restrict__ d, int n) {
  int i = blockIdx.x * 256 + threadIdx.x;
  if (i < n) d[i] = f2bf(s[i]);
}
__global__ __launch_bounds__(256) void cvt_bf16_T(const float* __restrict__ s,
                                                  unsigned short* __restrict__ d,
                                                  int K, int N) {
  int i = blockIdx.x * 256 + threadIdx.x;
  if (i < K * N) {
    int k = i / N, n = i % N;
    d[(size_t)n * K + k] = f2bf(s[i]);
  }
}

// ---- RoPE (in-place on bf16 buffer of shape [B*T, rowlen]) ----------------
__global__ __launch_bounds__(256) void rope_bf16(unsigned short* __restrict__ buf,
                                                 const float* __restrict__ fc,
                                                 const float* __restrict__ fs,
                                                 int rowlen, int total) {
  int i = blockIdx.x * 256 + threadIdx.x;
  if (i >= total) return;
  int half = rowlen >> 1;
  int pr = i % half;
  int bt = i / half;
  int t  = bt % T_;
  int p  = pr & (HD_ / 2 - 1);
  size_t base = (size_t)bt * rowlen + (size_t)pr * 2;
  float re = bf2f(buf[base]), im = bf2f(buf[base + 1]);
  float c = fc[t * (HD_ / 2) + p], s = fs[t * (HD_ / 2) + p];
  buf[base]     = f2bf(re * c - im * s);
  buf[base + 1] = f2bf(re * s + im * c);
}

// ---- LDS double-buffered GEMM: D = A(MxK) * W(KxN) given W^T(NxK) ---------
// Block tile 128x128, 8 waves of 32x64. MODE 0: bf16 out; 1: f32 out;
// 2: bf16 V^T scatter out[(b*N + n)*Tdim + t], row m = b*Tdim + t.
__device__ __forceinline__ void stage_gemm(const unsigned short* A,
                                           const unsigned short* BT,
                                           unsigned short* sA, unsigned short* sB,
                                           int bm, int bn, int K, int kc, int tid) {
#pragma unroll
  for (int u = 0; u < 2; ++u) {
    int j = tid * 2 + u;            // 0..511
    int row = j >> 2, chn = j & 3;  // 128 rows x 4 chunks of 8 halves
    async_ld16(sA + row * LDA + chn * 8, A + (size_t)(bm + row) * K + kc + chn * 8);
    async_ld16(sB + row * LDA + chn * 8, BT + (size_t)(bn + row) * K + kc + chn * 8);
  }
}

template <int MODE>
__global__ __launch_bounds__(256) void gemm_bf16_wmma(
    const unsigned short* __restrict__ A, const unsigned short* __restrict__ BT,
    void* __restrict__ D, int M, int N, int K, int Tdim) {
  __shared__ unsigned short sA[2][128 * LDA];
  __shared__ unsigned short sB[2][128 * LDA];
  int tid = threadIdx.x;
  int w = tid >> 5;
  int wm = w & 3, wn = w >> 2;                 // 4 (m) x 2 (n) waves
  int bm = blockIdx.y * 128, bn = blockIdx.x * 128;
  int tile_m = bm + wm * 32;
  int tile_n = bn + wn * 64;
  v8f acc[2][4];
#pragma unroll
  for (int s = 0; s < 2; ++s)
#pragma unroll
    for (int f = 0; f < 4; ++f)
#pragma unroll
      for (int r = 0; r < 8; ++r) acc[s][f][r] = 0.f;

  int nk = K >> 5;
  stage_gemm(A, BT, sA[0], sB[0], bm, bn, K, 0, tid);
  for (int i = 0; i < nk; ++i) {
    int buf = i & 1;
    wait_async();
    __syncthreads();
    if (i + 1 < nk)
      stage_gemm(A, BT, sA[buf ^ 1], sB[buf ^ 1], bm, bn, K, (i + 1) * 32, tid);
    const unsigned short* a0 = &sA[buf][(wm * 32) * LDA];
    v16bf af0 = load_a16x32(a0, LDA);
    v16bf af1 = load_a16x32(a0 + 16 * LDA, LDA);
#pragma unroll
    for (int f = 0; f < 4; ++f) {
      v16bf bfr = load_b32x16(&sB[buf][(wn * 64 + f * 16) * LDA], LDA);
      acc[0][f] = wmma_bf16(af0, bfr, acc[0][f]);
      acc[1][f] = wmma_bf16(af1, bfr, acc[1][f]);
    }
  }

  int lane = tid & 31;
  int cn = lane & 15, ch = lane >> 4;
#pragma unroll
  for (int s = 0; s < 2; ++s)
#pragma unroll
    for (int f = 0; f < 4; ++f) {
      int col = tile_n + f * 16 + cn;
#pragma unroll
      for (int r = 0; r < 8; ++r) {
        int row = tile_m + s * 16 + r + 8 * ch;
        float v = acc[s][f][r];
        if (MODE == 0) {
          ((unsigned short*)D)[(size_t)row * N + col] = f2bf(v);
        } else if (MODE == 1) {
          ((float*)D)[(size_t)row * N + col] = v;
        } else {
          int b = row / Tdim, t = row % Tdim;
          ((unsigned short*)D)[((size_t)(b * N + col)) * Tdim + t] = f2bf(v);
        }
      }
    }
}

// ---- Flash attention (causal, GQA rep=4), LDS-shared K/V tiles ------------
// K tile: 32 keys x 128 d (row stride LDK). V tile: 128 d x 32 t (stride LDA).
__device__ __forceinline__ void stage_attn(const unsigned short* kbase,
                                           const unsigned short* vbase,
                                           unsigned short* sK, unsigned short* sV,
                                           int kb, int tid) {
#pragma unroll
  for (int u = 0; u < 2; ++u) {
    int j = tid * 2 + u;              // 0..511
    int kr = j >> 4, kc = j & 15;     // 32 rows x 16 chunks of 8 halves
    async_ld16(sK + kr * LDK + kc * 8, kbase + (size_t)(kb + kr) * KV_ + kc * 8);
    int vr = j >> 2, vc = j & 3;      // 128 rows x 4 chunks of 8 halves
    async_ld16(sV + vr * LDA + vc * 8, vbase + (size_t)vr * T_ + kb + vc * 8);
  }
}

__global__ __launch_bounds__(256) void attn_wmma(const unsigned short* __restrict__ q,
                                                 const unsigned short* __restrict__ k,
                                                 const unsigned short* __restrict__ vT,
                                                 unsigned short* __restrict__ y) {
  __shared__ unsigned short sK[2][32 * LDK];
  __shared__ unsigned short sV[2][128 * LDA];
  __shared__ unsigned short plds[8 * 512];
  const float SCALE = 0.08838834764831845f;   // 1/sqrt(128)
  int tid = threadIdx.x;
  int w = tid >> 5, lane = tid & 31;
  int cn = lane & 15, ch = lane >> 4;
  int b = blockIdx.z, h = blockIdx.y;
  int t0b = blockIdx.x * 128;
  int t0 = t0b + w * 16;
  int kvh = h >> 2;

  const unsigned short* qbase = q + ((size_t)(b * T_ + t0)) * C_ + h * HD_;
  const unsigned short* kbase = k + ((size_t)b * T_) * KV_ + kvh * HD_;
  const unsigned short* vbase = vT + ((size_t)(b * KV_ + kvh * HD_)) * T_;

  v16bf qf[4];
#pragma unroll
  for (int c = 0; c < 4; ++c) qf[c] = load_a16x32(qbase + c * 32, C_);

  v8f o[8];
  float m_i[8], l_i[8];
#pragma unroll
  for (int r = 0; r < 8; ++r) { m_i[r] = -3.0e38f; l_i[r] = 0.f; }
#pragma unroll
  for (int f = 0; f < 8; ++f)
#pragma unroll
    for (int r = 0; r < 8; ++r) o[f][r] = 0.f;

  int nsteps = (t0b + 128) >> 5;   // uniform per block
  stage_attn(kbase, vbase, sK[0], sV[0], 0, tid);
  for (int it = 0; it < nsteps; ++it) {
    int kb = it * 32;
    int buf = it & 1;
    wait_async();
    __syncthreads();
    if (it + 1 < nsteps)
      stage_attn(kbase, vbase, sK[buf ^ 1], sV[buf ^ 1], kb + 32, tid);

    if (kb < t0 + 16) {
      v8f s0, s1;
#pragma unroll
      for (int r = 0; r < 8; ++r) { s0[r] = 0.f; s1[r] = 0.f; }
      const unsigned short* kt = &sK[buf][0];
#pragma unroll
      for (int c = 0; c < 4; ++c) {
        v16bf kf0 = load_b32x16(kt + c * 32, LDK);
        s0 = wmma_bf16(qf[c], kf0, s0);
        v16bf kf1 = load_b32x16(kt + 16 * LDK + c * 32, LDK);
        s1 = wmma_bf16(qf[c], kf1, s1);
      }
      float p0[8], p1[8], alpha[8];
#pragma unroll
      for (int r = 0; r < 8; ++r) {
        int qi = t0 + r + 8 * ch;
        float a0 = (kb + cn <= qi) ? s0[r] * SCALE : -3.0e38f;
        float a1 = (kb + 16 + cn <= qi) ? s1[r] * SCALE : -3.0e38f;
        float rm = fmaxf(a0, a1);
        rm = fmaxf(rm, __shfl_xor(rm, 1));
        rm = fmaxf(rm, __shfl_xor(rm, 2));
        rm = fmaxf(rm, __shfl_xor(rm, 4));
        rm = fmaxf(rm, __shfl_xor(rm, 8));
        float mn = fmaxf(m_i[r], rm);
        alpha[r] = __expf(m_i[r] - mn);
        m_i[r] = mn;
        p0[r] = __expf(a0 - mn);
        p1[r] = __expf(a1 - mn);
        float rs = p0[r] + p1[r];
        rs += __shfl_xor(rs, 1);
        rs += __shfl_xor(rs, 2);
        rs += __shfl_xor(rs, 4);
        rs += __shfl_xor(rs, 8);
        l_i[r] = l_i[r] * alpha[r] + rs;
      }
#pragma unroll
      for (int f = 0; f < 8; ++f)
#pragma unroll
        for (int r = 0; r < 8; ++r) o[f][r] *= alpha[r];

      unsigned short* pl = plds + w * 512;
#pragma unroll
      for (int r = 0; r < 8; ++r) {
        int row = r + 8 * ch;
        pl[row * 32 + cn]      = f2bf(p0[r]);
        pl[row * 32 + 16 + cn] = f2bf(p1[r]);
      }
      v16bf pf = load_a16x32(pl, 32);
#pragma unroll
      for (int f = 0; f < 8; ++f) {
        v16bf vf = load_b32x16(&sV[buf][(f * 16) * LDA], LDA);
        o[f] = wmma_bf16(pf, vf, o[f]);
      }
    }
  }

  unsigned short* ybase = y + ((size_t)(b * T_ + t0)) * C_ + h * HD_;
#pragma unroll
  for (int r = 0; r < 8; ++r) {
    float inv = 1.f / l_i[r];
    int row = r + 8 * ch;
#pragma unroll
    for (int f = 0; f < 8; ++f)
      ybase[(size_t)row * C_ + f * 16 + cn] = f2bf(o[f][r] * inv);
  }
}

// ---- host orchestration ----------------------------------------------------
extern "C" void kernel_launch(void* const* d_in, const int* in_sizes, int n_in,
                              void* d_out, int out_size, void* d_ws, size_t ws_size,
                              hipStream_t stream) {
  (void)in_sizes; (void)n_in; (void)out_size; (void)ws_size;
  const float* x  = (const float*)d_in[0];
  const float* Wq = (const float*)d_in[1];
  const float* Wk = (const float*)d_in[2];
  const float* Wv = (const float*)d_in[3];
  const float* Wo = (const float*)d_in[4];
  const float* fc = (const float*)d_in[5];
  const float* fs = (const float*)d_in[6];
  float* out = (float*)d_out;

  unsigned short* ws = (unsigned short*)d_ws;
  unsigned short* xb   = ws;                              // M_*C_
  unsigned short* qb   = xb   + (size_t)M_ * C_;          // M_*C_
  unsigned short* yb   = qb   + (size_t)M_ * C_;          // M_*C_
  unsigned short* WqT  = yb   + (size_t)M_ * C_;          // C_*C_
  unsigned short* WoT  = WqT  + (size_t)C_ * C_;          // C_*C_
  unsigned short* WkT  = WoT  + (size_t)C_ * C_;          // KV_*C_
  unsigned short* WvT  = WkT  + (size_t)KV_ * C_;         // KV_*C_
  unsigned short* kbuf = WvT  + (size_t)KV_ * C_;         // M_*KV_
  unsigned short* vT   = kbuf + (size_t)M_ * KV_;         // B_*KV_*T_

  {
    int n = M_ * C_;
    cvt_bf16<<<(n + 255) / 256, 256, 0, stream>>>(x, xb, n);
    n = C_ * C_;
    cvt_bf16_T<<<(n + 255) / 256, 256, 0, stream>>>(Wq, WqT, C_, C_);
    cvt_bf16_T<<<(n + 255) / 256, 256, 0, stream>>>(Wo, WoT, C_, C_);
    n = C_ * KV_;
    cvt_bf16_T<<<(n + 255) / 256, 256, 0, stream>>>(Wk, WkT, C_, KV_);
    cvt_bf16_T<<<(n + 255) / 256, 256, 0, stream>>>(Wv, WvT, C_, KV_);
  }
  gemm_bf16_wmma<0><<<dim3(C_ / 128, M_ / 128), 256, 0, stream>>>(xb, WqT, qb, M_, C_, C_, 0);
  gemm_bf16_wmma<0><<<dim3(KV_ / 128, M_ / 128), 256, 0, stream>>>(xb, WkT, kbuf, M_, KV_, C_, 0);
  gemm_bf16_wmma<2><<<dim3(KV_ / 128, M_ / 128), 256, 0, stream>>>(xb, WvT, vT, M_, KV_, C_, T_);
  {
    int n = M_ * C_ / 2;
    rope_bf16<<<(n + 255) / 256, 256, 0, stream>>>(qb, fc, fs, C_, n);
    n = M_ * KV_ / 2;
    rope_bf16<<<(n + 255) / 256, 256, 0, stream>>>(kbuf, fc, fs, KV_, n);
  }
  attn_wmma<<<dim3(T_ / 128, NH_, B_), 256, 0, stream>>>(qb, kbuf, vT, yb);
  gemm_bf16_wmma<1><<<dim3(C_ / 128, M_ / 128), 256, 0, stream>>>(yb, WoT, out, M_, C_, C_, 0);
}
